// GRU_15839839387723
// MI455X (gfx1250) — matrix operand order repaired and checked by reading, not compiled, not run on hardware
//
#include <hip/hip_runtime.h>

// ---------------------------------------------------------------------------
// GRU forward, MI455X (gfx1250, wave32, WMMA + async global->LDS).
// Phase 1: x_proj = x @ w_ih^T + b_ih   (bf16 WMMA GEMM, f32 accum,
//          A-tile staged via GLOBAL_LOAD_ASYNC_TO_LDS_B128, double-buffered,
//          branchless modulo-prefetch so the mainloop is one straight block)
// Phase 2: persistent recurrent kernel, device-wide barrier per timestep;
//          w_hh resident in LDS, h-tile async-copied to LDS each step,
//          96 v_wmma_f32_16x16x32_bf16 per wave per step, h kept in regs.
// ---------------------------------------------------------------------------

typedef __bf16 bf16;
typedef __attribute__((ext_vector_type(16))) __bf16 v16bf;
typedef __attribute__((ext_vector_type(8)))  __bf16 v8bf;
typedef __attribute__((ext_vector_type(8)))  float  v8f;

static constexpr int T  = 512;
static constexpr int B  = 64;
static constexpr int I  = 1024;
static constexpr int H  = 1024;
static constexpr int H3 = 3 * H;      // 3072
static constexpr int M  = T * B;      // 32768 rows of x_proj / out

// ---- CDNA5 async global->LDS helpers (ASYNCcnt-tracked DMA) ---------------

__device__ __forceinline__ void async_load_b128(unsigned lds_off,
                                                const void* gaddr) {
  // GLOBAL_LOAD_ASYNC_TO_LDS_B128: vdst = LDS byte address, vaddr = 64b GVA.
  asm volatile("global_load_async_to_lds_b128 %0, %1, off"
               :: "v"(lds_off), "v"(gaddr) : "memory");
}
__device__ __forceinline__ void wait_async_le1() {
  asm volatile("s_wait_asynccnt 0x1" ::: "memory");
}
__device__ __forceinline__ void wait_async_0() {
  asm volatile("s_wait_asynccnt 0x0" ::: "memory");
}
// Low 32 bits of a generic pointer to __shared__ = LDS byte offset.
__device__ __forceinline__ unsigned lds_addr_of(const void* p) {
  return (unsigned)(uintptr_t)p;
}

// ---- math helpers ----------------------------------------------------------

__device__ __forceinline__ bf16 f2bf(float f) {
  unsigned u = __builtin_bit_cast(unsigned, f);
  unsigned r = u + 0x7FFFu + ((u >> 16) & 1u);   // round-to-nearest-even
  unsigned short h = (unsigned short)(r >> 16);
  return __builtin_bit_cast(bf16, h);
}
__device__ __forceinline__ float sigmoid_f(float x) {
  return 1.0f / (1.0f + __expf(-x));
}
__device__ __forceinline__ float tanh_f(float x) {
  return 1.0f - 2.0f / (__expf(2.0f * x) + 1.0f);   // monotone, no NaN at +-inf
}

// Build a 16x32 bf16 A-fragment in contiguous K order (same order as the B loads):
//   lane L: row m = L&15; K = [(L>>4)*16 .. +7] in elems 0..7,
//                         K = [(L>>4)*16+8 .. +15] in elems 8..15.
__device__ __forceinline__ v16bf make_a_frag(const bf16* rowPtr, int kA) {
  v8bf lo = *(const v8bf*)(rowPtr + kA);
  v8bf hi = *(const v8bf*)(rowPtr + kA + 8);
  v16bf a;
#pragma unroll
  for (int e = 0; e < 8; ++e) { a[e] = lo[e]; a[e + 8] = hi[e]; }
  return a;
}

// ---- small utility kernels -------------------------------------------------

__global__ void init_sync_kernel(unsigned* sync) {
  sync[0] = 0u;   // arrival counter
  sync[1] = 0u;   // generation
}

__global__ void cvt_f32_bf16(const float* __restrict__ src,
                             bf16* __restrict__ dst, int n) {
  int i = blockIdx.x * blockDim.x + threadIdx.x;
  int stride = gridDim.x * blockDim.x;
  for (; i < n; i += stride) dst[i] = f2bf(src[i]);
}

// ---- phase 1: x_proj GEMM --------------------------------------------------
// out[m, g] = sum_k x[m,k] * w_ih[g,k] + b_ih[g]
// WG = 256 threads = 8 waves (2 in M x 4 in N); each wave computes 32x64.
// A tile (64 rows x 32 K) async-staged into LDS, double-buffered.

static constexpr int AP   = 36;           // padded A row (elems): 72 B
static constexpr int ABUF = 64 * AP;      // one A buffer (elems)

__global__ void __launch_bounds__(256)
xproj_gemm(const bf16* __restrict__ Xb,     // [M, I]  row-major, K contiguous
           const bf16* __restrict__ Wb,     // [H3, I] row-major (row = out col)
           const float* __restrict__ bias,  // [H3]
           float* __restrict__ out) {       // [M, H3]
  __shared__ bf16 sA[2 * ABUF];            // 9216 B

  const int tid  = threadIdx.x;
  const int lane = tid & 31;
  const int wave = tid >> 5;
  const int wm   = wave & 1;
  const int mBase = blockIdx.y * 64;
  const int n0    = blockIdx.x * 256 + (wave >> 1) * 64;

  // --- async A staging map: thread -> (row, 16B segment) --------------------
  const int ldRow = tid >> 2;              // 0..63
  const int ldSeg = tid & 3;               // 0..3 (16 B each, 32 elems/row)
  const bf16* gA = Xb + (size_t)(mBase + ldRow) * I + ldSeg * 8;
  const unsigned ldsBase = lds_addr_of(sA);
  const unsigned ldsSlot = ldRow * (AP * 2) + ldSeg * 16;  // byte offset

  async_load_b128(ldsBase + ldsSlot, gA);  // prologue: chunk 0 -> buf 0

  const int am = lane & 15;                // A row within 16-row tile
  const int kA = (lane >> 4) * 16;         // A K-subchunk base (elems)
  const int bn = lane & 15;                // B col within tile
  const int kB = (lane >> 4) * 16;         // B K-subchunk base (elems)
  const int rhalf = (lane >> 4) * 8;       // C/D row half

  v8f acc[2][4] = {};

  const bf16* bRow[4];
#pragma unroll
  for (int j = 0; j < 4; ++j)
    bRow[j] = Wb + (size_t)(n0 + j * 16 + bn) * I;

  int buf = 0;
  for (int k0 = 0; k0 < I; k0 += 32, buf ^= 1) {
    // Branchless prefetch of the next chunk into the other buffer. On the
    // final iteration this re-fetches chunk 0 into the dead buffer (never
    // read; drained by s_endpgm's implicit wait-idle) -- keeps the loop body
    // a single straight-line block with a fixed s_wait_asynccnt 0x1.
    const int knext = (k0 + 32) & (I - 1);
    async_load_b128(ldsBase + (buf ? 0u : (unsigned)(ABUF * 2)) + ldsSlot,
                    gA + knext);
    wait_async_le1();  // current chunk complete (async loads finish in order)
    __syncthreads();

    const bf16* aBuf  = sA + buf * ABUF;
    const bf16* aRow0 = aBuf + (size_t)(wm * 32 + am) * AP;
    const bf16* aRow1 = aRow0 + (size_t)16 * AP;
    v16bf a0 = make_a_frag(aRow0, kA);     // ds_load_b128 pairs
    v16bf a1 = make_a_frag(aRow1, kA);

    v16bf b[4];
#pragma unroll
    for (int j = 0; j < 4; ++j)
      b[j] = *(const v16bf*)(bRow[j] + k0 + kB);   // 32 B contiguous (L2-hot)

#pragma unroll
    for (int j = 0; j < 4; ++j) {
      acc[0][j] = __builtin_amdgcn_wmma_f32_16x16x32_bf16(
          false, a0, false, b[j], (short)0, acc[0][j], false, false);
      acc[1][j] = __builtin_amdgcn_wmma_f32_16x16x32_bf16(
          false, a1, false, b[j], (short)0, acc[1][j], false, false);
    }
    __syncthreads();     // all reads done before next overwrite
  }

#pragma unroll
  for (int j = 0; j < 4; ++j) {
    const int col = n0 + j * 16 + bn;
    const float bb = bias[col];
#pragma unroll
    for (int i = 0; i < 2; ++i)
#pragma unroll
      for (int v = 0; v < 8; ++v) {
        const int row = mBase + wm * 32 + i * 16 + v + rhalf;
        out[(size_t)row * H3 + col] = acc[i][j][v] + bb;
      }
  }
}

// ---- phase 2: persistent recurrence ---------------------------------------
// 256 WGs x 32 threads. WG owns batch-rows [r0,r0+16) x hidden-cols [c0,c0+16).
// LDS: w_hh tile (48 x 1040 bf16, loaded once) + h tile (16 x 1040, per step).

static constexpr int LROW = H + 16;       // padded LDS row (elems), 2080 B

__device__ __forceinline__ void grid_barrier(unsigned* sync, unsigned nwg) {
  __threadfence();
  if (threadIdx.x == 0) {
    volatile unsigned* genp = sync + 1;
    const unsigned g = *genp;
    if (atomicAdd(sync, 1u) == nwg - 1u) {
      atomicExch(sync, 0u);
      __threadfence();
      atomicAdd((unsigned*)(sync + 1), 1u);
    } else {
      while (*genp == g) __builtin_amdgcn_s_sleep(2);
    }
  }
  __threadfence();
  __syncthreads();
}

__global__ void __launch_bounds__(32)
gru_recurrent(const float* __restrict__ xproj,  // [M, H3] (includes b_ih)
              const bf16*  __restrict__ whh,    // [H3, H] bf16
              const float* __restrict__ bhh,    // [H3]
              const float* __restrict__ h0,     // [B, H] f32
              bf16* __restrict__ hb0,           // [B, H] bf16 ping
              bf16* __restrict__ hb1,           // [B, H] bf16 pong
              float* __restrict__ out,          // [M, H]
              unsigned* __restrict__ sync) {
  extern __shared__ bf16 smem[];                // [48+16][LROW]
  bf16* sW = smem;                              // w_hh tile
  bf16* sH = smem + 48 * LROW;                  // h tile (per step)

  const int lane = threadIdx.x;
  const int r0 = (blockIdx.x >> 6) * 16;        // batch tile
  const int c0 = (blockIdx.x & 63) * 16;        // hidden-col tile
  const unsigned nwg = gridDim.x;
  const unsigned sHbase = lds_addr_of(sH);

  // Stage w_hh tile into LDS once (reused for all 512 steps).
  for (int r = 0; r < 48; ++r) {
    const int gi = r >> 4, nl = r & 15;
    const bf16* src = whh + (size_t)(gi * H + c0 + nl) * H;
    bf16* dst = sW + r * LROW;
    for (int e = lane * 8; e < H; e += 32 * 8)
      *(v8bf*)(dst + e) = *(const v8bf*)(src + e);
  }
  __syncthreads();

  const int am = lane & 15;
  const int kA = (lane >> 4) * 16;
  const int bn = lane & 15;
  const int kB = (lane >> 4) * 16;
  const int rhalf = (lane >> 4) * 8;
  const int col = c0 + bn;

  const float bhr = bhh[col], bhz = bhh[H + col], bhn = bhh[2 * H + col];

  // h_old for this tile lives in registers across the whole time loop.
  float hold[8];
#pragma unroll
  for (int v = 0; v < 8; ++v)
    hold[v] = h0[(size_t)(r0 + v + rhalf) * H + col];

  for (int t = 0; t < T; ++t) {
    const bf16* hprev = (t & 1) ? hb1 : hb0;
    bf16*       hcur  = (t & 1) ? hb0 : hb1;

    // Burst async copy of the 16 x 1024 h tile into LDS (coalesced DMA).
#pragma unroll
    for (int row = 0; row < 16; ++row) {
      const bf16* gsrc = hprev + (size_t)(r0 + row) * H + lane * 8;
      const unsigned ldst = sHbase + row * (LROW * 2) + lane * 16;
#pragma unroll
      for (int q = 0; q < 4; ++q)
        async_load_b128(ldst + q * 512, gsrc + q * 256);
    }

    // Prefetch this step's x_proj operands under the DMA/WMMA shadow.
#pragma unroll
    for (int v = 0; v < 8; ++v) {
      const size_t mrow = (size_t)t * B + (r0 + v + rhalf);
      __builtin_prefetch(&xproj[mrow * H3 + col], 0, 0);
      __builtin_prefetch(&xproj[mrow * H3 + H + col], 0, 0);
      __builtin_prefetch(&xproj[mrow * H3 + 2 * H + col], 0, 0);
    }

    wait_async_0();   // h tile resident in LDS (single-wave WG: no barrier)

    v8f accr = {}, accz = {}, accn = {};
    const bf16* aRow = sH + (size_t)am * LROW;
    for (int k0 = 0; k0 < H; k0 += 32) {
      v16bf a  = make_a_frag(aRow + k0, kA);
      v16bf br = *(const v16bf*)(sW + (size_t)( 0 + bn) * LROW + k0 + kB);
      v16bf bz = *(const v16bf*)(sW + (size_t)(16 + bn) * LROW + k0 + kB);
      v16bf bm = *(const v16bf*)(sW + (size_t)(32 + bn) * LROW + k0 + kB);
      accr = __builtin_amdgcn_wmma_f32_16x16x32_bf16(
          false, a, false, br, (short)0, accr, false, false);
      accz = __builtin_amdgcn_wmma_f32_16x16x32_bf16(
          false, a, false, bz, (short)0, accz, false, false);
      accn = __builtin_amdgcn_wmma_f32_16x16x32_bf16(
          false, a, false, bm, (short)0, accn, false, false);
    }

    // Fused GRU gate math on the 16x16 accumulator tile.
#pragma unroll
    for (int v = 0; v < 8; ++v) {
      const int b = r0 + v + rhalf;
      const size_t mrow = (size_t)t * B + b;
      const float xr = xproj[mrow * H3 + col];
      const float xz = xproj[mrow * H3 + H + col];
      const float xn = xproj[mrow * H3 + 2 * H + col];
      const float r = sigmoid_f(xr + accr[v] + bhr);
      const float z = sigmoid_f(xz + accz[v] + bhz);
      const float n = tanh_f(xn + r * (accn[v] + bhn));
      const float hnew = (1.0f - z) * n + z * hold[v];
      hold[v] = hnew;
      out[mrow * H + col] = hnew;
      hcur[(size_t)b * H + col] = f2bf(hnew);
    }

    grid_barrier(sync, nwg);   // publish h_t before anyone starts step t+1
  }
}

// ---- host ------------------------------------------------------------------

extern "C" void kernel_launch(void* const* d_in, const int* in_sizes, int n_in,
                              void* d_out, int out_size, void* d_ws, size_t ws_size,
                              hipStream_t stream) {
  (void)in_sizes; (void)n_in; (void)out_size; (void)ws_size;

  const float* x    = (const float*)d_in[0];   // [T,B,I]
  const float* h0   = (const float*)d_in[1];   // [1,B,H]
  const float* w_ih = (const float*)d_in[2];   // [3H,I]
  const float* w_hh = (const float*)d_in[3];   // [3H,H]
  const float* b_ih = (const float*)d_in[4];   // [3H]
  const float* b_hh = (const float*)d_in[5];   // [3H]
  float* out = (float*)d_out;                  // [T*B, H]

  char* ws = (char*)d_ws;
  size_t off = 0;
  unsigned* sync = (unsigned*)(ws + off); off += 256;
  bf16* wih_b = (bf16*)(ws + off);  off += (size_t)H3 * I * sizeof(bf16);
  bf16* whh_b = (bf16*)(ws + off);  off += (size_t)H3 * H * sizeof(bf16);
  bf16* x_b   = (bf16*)(ws + off);  off += (size_t)M * I * sizeof(bf16);
  bf16* hb0   = (bf16*)(ws + off);  off += (size_t)B * H * sizeof(bf16);
  bf16* hb1   = (bf16*)(ws + off);  off += (size_t)B * H * sizeof(bf16);
  float* xproj = (float*)(ws + off); off += (size_t)M * H3 * sizeof(float);

  init_sync_kernel<<<1, 1, 0, stream>>>(sync);
  cvt_f32_bf16<<<1024, 256, 0, stream>>>(w_ih, wih_b, H3 * I);
  cvt_f32_bf16<<<1024, 256, 0, stream>>>(w_hh, whh_b, H3 * H);
  cvt_f32_bf16<<<4096, 256, 0, stream>>>(x, x_b, M * I);

  dim3 ggrid(H3 / 256, M / 64);                // 12 x 512 WGs
  xproj_gemm<<<ggrid, 256, 0, stream>>>(x_b, wih_b, b_ih, xproj);

  const int nwg = (B / 16) * (H / 16);         // 256 single-wave WGs
  const size_t lds = (size_t)(48 + 16) * LROW * sizeof(bf16);  // 133,120 B
  gru_recurrent<<<nwg, 32, lds, stream>>>(xproj, whh_b, b_hh, h0,
                                          hb0, hb1, out, sync);
}
